// QuantizeLinear_17214228922556
// MI455X (gfx1250) — matrix operand order repaired and checked by reading, not compiled, not run on hardware
//
#include <hip/hip_runtime.h>

typedef __attribute__((ext_vector_type(4)))  _Float16 v4h;
typedef __attribute__((ext_vector_type(8)))  _Float16 v8h;
typedef __attribute__((ext_vector_type(16))) _Float16 v16h;
typedef __attribute__((ext_vector_type(8)))  float    v8f;

#define D_DIM 4096
#define M_DIM 16384   /* B*S = 4*4096 */
#define LN_EPS 1e-5f

#define BM 128
#define BN 128
#define BK 32
#define PAD 8
#define KITER (D_DIM / BK)   /* 128 */

#if defined(__has_builtin)
#  if __has_builtin(__builtin_amdgcn_global_load_async_to_lds_b128)
#    define HAVE_ASYNC_LDS 1
#  endif
#  if __has_builtin(__builtin_amdgcn_s_wait_asynccnt)
#    define HAVE_WAIT_ASYNC 1
#  endif
#endif

typedef int v4i_g __attribute__((vector_size(16)));
#ifdef HAVE_ASYNC_LDS
typedef __attribute__((address_space(1))) v4i_g* gasync_ptr;
typedef __attribute__((address_space(3))) v4i_g* lasync_ptr;
#endif

// 16-byte global -> LDS copy; async (ASYNCcnt) when the toolchain exposes it.
__device__ __forceinline__ void copy16_to_lds(const _Float16* gsrc, _Float16* lds) {
#ifdef HAVE_ASYNC_LDS
  __builtin_amdgcn_global_load_async_to_lds_b128(
      (gasync_ptr)(unsigned long long)(__UINTPTR_TYPE__)gsrc,
      (lasync_ptr)(unsigned int)(__UINTPTR_TYPE__)lds,
      /*offset=*/0, /*cpol=*/0);
#else
  *(v8h*)lds = *(const v8h*)gsrc;
#endif
}

__device__ __forceinline__ void wait_async_copies() {
#if defined(HAVE_WAIT_ASYNC)
  __builtin_amdgcn_s_wait_asynccnt(0);
#elif defined(HAVE_ASYNC_LDS)
  asm volatile("s_wait_asynccnt 0x0" ::: "memory");
#endif
}

// ---------------- prep: A = (x * g) -> f16, row-major [M, K] ----------------
__global__ __launch_bounds__(256)
void scale_cast_kernel(const float* __restrict__ x,
                       const float* __restrict__ g,
                       _Float16* __restrict__ A) {
  size_t idx = ((size_t)blockIdx.x * 256 + threadIdx.x) * 4;
  int k = (int)(idx & (D_DIM - 1));
  float4 xv = *(const float4*)(x + idx);
  float4 gv = *(const float4*)(g + k);
  v4h o;
  o.x = (_Float16)(xv.x * gv.x);
  o.y = (_Float16)(xv.y * gv.y);
  o.z = (_Float16)(xv.z * gv.z);
  o.w = (_Float16)(xv.w * gv.w);
  *(v4h*)(A + idx) = o;
}

// ---------------- prep: Wf = sign(W) -> f16, row-major [N, K] ---------------
__global__ __launch_bounds__(256)
void sign_cast_kernel(const float* __restrict__ w,
                      _Float16* __restrict__ Wf) {
  size_t idx = ((size_t)blockIdx.x * 256 + threadIdx.x) * 4;
  float4 wv = *(const float4*)(w + idx);
  v4h o;
  o.x = (_Float16)((wv.x > 0.f) ? 1.f : ((wv.x < 0.f) ? -1.f : 0.f));
  o.y = (_Float16)((wv.y > 0.f) ? 1.f : ((wv.y < 0.f) ? -1.f : 0.f));
  o.z = (_Float16)((wv.z > 0.f) ? 1.f : ((wv.z < 0.f) ? -1.f : 0.f));
  o.w = (_Float16)((wv.w > 0.f) ? 1.f : ((wv.w < 0.f) ? -1.f : 0.f));
  *(v4h*)(Wf + idx) = o;
}

// ---------------- WMMA GEMM: Y[M,N] = A[M,K] * Wf[N,K]^T, +bias, *h ---------
__global__ __launch_bounds__(256)
void bitlinear_gemm_kernel(const _Float16* __restrict__ A,   // [M,K] f16
                           const _Float16* __restrict__ Wt,  // [N,K] f16 (+-1)
                           const float* __restrict__ bias,   // [N]
                           const float* __restrict__ h,      // [N]
                           float* __restrict__ Y) {          // [M,N] f32
  __shared__ _Float16 As[2][BM][BK + PAD];
  __shared__ _Float16 Bs[2][BN][BK + PAD];

  const int tid  = threadIdx.x;
  const int lane = tid & 31;
  const int wave = tid >> 5;      // 0..7
  const int wm   = wave & 3;      // 4 waves along M (32 rows each)
  const int wn   = wave >> 2;     // 2 waves along N (64 cols each)

  const int gm0 = blockIdx.y * BM;
  const int gn0 = blockIdx.x * BN;

  const int half = lane >> 4;     // 0 or 1
  const int l16  = lane & 15;

  // Staging geometry: tile = 128 rows x 32 halves = 512 chunks of 16B,
  // 256 threads -> 2 chunks each, per tile.
  const int c0 = tid;
  const int r0 = c0 >> 2,  cc0 = (c0 & 3) * 8;
  const int c1 = tid + 256;
  const int r1 = c1 >> 2,  cc1 = (c1 & 3) * 8;

  v8f acc[2][4] = {};

  // Prologue: stage k-tile 0 into buffer 0.
  {
    copy16_to_lds(A  + (size_t)(gm0 + r0) * D_DIM + cc0, &As[0][r0][cc0]);
    copy16_to_lds(Wt + (size_t)(gn0 + r0) * D_DIM + cc0, &Bs[0][r0][cc0]);
    copy16_to_lds(A  + (size_t)(gm0 + r1) * D_DIM + cc1, &As[0][r1][cc1]);
    copy16_to_lds(Wt + (size_t)(gn0 + r1) * D_DIM + cc1, &Bs[0][r1][cc1]);
  }
  wait_async_copies();
  __syncthreads();

  for (int it = 0; it < KITER; ++it) {
    const int cur = it & 1;

    // Stage next k-tile into the other buffer (overlaps with WMMA below).
    if (it + 1 < KITER) {
      const int nb = cur ^ 1;
      const int kk = (it + 1) * BK;
      copy16_to_lds(A  + (size_t)(gm0 + r0) * D_DIM + kk + cc0, &As[nb][r0][cc0]);
      copy16_to_lds(Wt + (size_t)(gn0 + r0) * D_DIM + kk + cc0, &Bs[nb][r0][cc0]);
      copy16_to_lds(A  + (size_t)(gm0 + r1) * D_DIM + kk + cc1, &As[nb][r1][cc1]);
      copy16_to_lds(Wt + (size_t)(gn0 + r1) * D_DIM + kk + cc1, &Bs[nb][r1][cc1]);
    }

    // A fragments: lane half 0 -> K {0..7,16..23}; half 1 -> K {8..15,24..31}
    v16h afrag[2];
    #pragma unroll
    for (int i = 0; i < 2; ++i) {
      const _Float16* ap = &As[cur][wm * 32 + i * 16 + l16][half * 8];
      v8h lo = *(const v8h*)ap;
      v8h hi = *(const v8h*)(ap + 16);
      afrag[i] = __builtin_shufflevector(lo, hi, 0,1,2,3,4,5,6,7,8,9,10,11,12,13,14,15);
    }
    // B fragments: lane = column; half 0 -> K {0..7,16..23}; half 1 -> K {8..15,24..31}
    v16h bfrag[4];
    #pragma unroll
    for (int j = 0; j < 4; ++j) {
      const _Float16* bp = &Bs[cur][wn * 64 + j * 16 + l16][half * 8];
      v8h lo = *(const v8h*)bp;
      v8h hi = *(const v8h*)(bp + 16);
      bfrag[j] = __builtin_shufflevector(lo, hi, 0,1,2,3,4,5,6,7,8,9,10,11,12,13,14,15);
    }
    #pragma unroll
    for (int i = 0; i < 2; ++i)
      #pragma unroll
      for (int j = 0; j < 4; ++j)
        acc[i][j] = __builtin_amdgcn_wmma_f32_16x16x32_f16(
            /*neg_a=*/false, afrag[i], /*neg_b=*/false, bfrag[j],
            /*c_mod=*/(short)0, acc[i][j], /*reuse_a=*/false, /*reuse_b=*/false);

    wait_async_copies();   // my async copies into the next buffer are done
    __syncthreads();       // everyone done reading cur + everyone's copies done
  }

  // Epilogue: C layout -> row = base + v + 8*half, col = base + l16
  #pragma unroll
  for (int i = 0; i < 2; ++i) {
    const int rbase = gm0 + wm * 32 + i * 16 + half * 8;
    #pragma unroll
    for (int j = 0; j < 4; ++j) {
      const int col = gn0 + wn * 64 + j * 16 + l16;
      const float bcol = bias[col];
      const float hcol = h[col];
      #pragma unroll
      for (int v = 0; v < 8; ++v) {
        Y[(size_t)(rbase + v) * D_DIM + col] = (acc[i][j][v] + bcol) * hcol;
      }
    }
  }
}

// ---------------- LayerNorm over out-dim, in place on Y --------------------
__global__ __launch_bounds__(256)
void layernorm_kernel(float* __restrict__ Y,
                      const float* __restrict__ gamma,
                      const float* __restrict__ beta) {
  float* y = Y + (size_t)blockIdx.x * D_DIM;
  const int tid = threadIdx.x;

  float4 vals[4];
  float s = 0.f, s2 = 0.f;
  #pragma unroll
  for (int i = 0; i < 4; ++i) {
    int idx = (tid + 256 * i) * 4;
    float4 v = *(const float4*)(y + idx);
    vals[i] = v;
    s  += v.x + v.y + v.z + v.w;
    s2 += v.x * v.x + v.y * v.y + v.z * v.z + v.w * v.w;
  }
  // wave32 reduce
  #pragma unroll
  for (int off = 16; off > 0; off >>= 1) {
    s  += __shfl_down(s,  off, 32);
    s2 += __shfl_down(s2, off, 32);
  }
  __shared__ float ss[8], ss2[8];
  if ((tid & 31) == 0) { ss[tid >> 5] = s; ss2[tid >> 5] = s2; }
  __syncthreads();
  s = 0.f; s2 = 0.f;
  #pragma unroll
  for (int w = 0; w < 8; ++w) { s += ss[w]; s2 += ss2[w]; }

  const float inv_d = 1.0f / (float)D_DIM;
  const float mu  = s * inv_d;
  const float var = s2 * inv_d - mu * mu;
  const float rs  = rsqrtf(var + LN_EPS);

  #pragma unroll
  for (int i = 0; i < 4; ++i) {
    int idx = (tid + 256 * i) * 4;
    float4 v  = vals[i];
    float4 gv = *(const float4*)(gamma + idx);
    float4 bv = *(const float4*)(beta + idx);
    float4 o;
    o.x = (v.x - mu) * rs * gv.x + bv.x;
    o.y = (v.y - mu) * rs * gv.y + bv.y;
    o.z = (v.z - mu) * rs * gv.z + bv.z;
    o.w = (v.w - mu) * rs * gv.w + bv.w;
    *(float4*)(y + idx) = o;
  }
}

extern "C" void kernel_launch(void* const* d_in, const int* in_sizes, int n_in,
                              void* d_out, int out_size, void* d_ws, size_t ws_size,
                              hipStream_t stream) {
  const float* x        = (const float*)d_in[0];
  const float* weight   = (const float*)d_in[1];
  const float* bias     = (const float*)d_in[2];
  const float* g        = (const float*)d_in[3];
  const float* h        = (const float*)d_in[4];
  const float* ln_gamma = (const float*)d_in[5];
  const float* ln_beta  = (const float*)d_in[6];
  float* out = (float*)d_out;

  _Float16* Af = (_Float16*)d_ws;                                  // 134 MB
  _Float16* Wf = (_Float16*)((char*)d_ws + (size_t)M_DIM * D_DIM * sizeof(_Float16));

  scale_cast_kernel<<<(M_DIM * (size_t)D_DIM) / 4 / 256, 256, 0, stream>>>(x, g, Af);
  sign_cast_kernel<<<((size_t)D_DIM * D_DIM) / 4 / 256, 256, 0, stream>>>(weight, Wf);

  dim3 grid(D_DIM / BN, M_DIM / BM);
  bitlinear_gemm_kernel<<<grid, 256, 0, stream>>>(Af, Wf, bias, h, out);

  layernorm_kernel<<<M_DIM, 256, 0, stream>>>(out, ln_gamma, ln_beta);
}
